// TwoLevelTransformerEncoderLayer_84086869721177
// MI455X (gfx1250) — compile-verified, matrix operand order checked by reading, not run on hardware
//
#include <hip/hip_runtime.h>
#include <cmath>

// ---------------- problem constants ----------------
#define BB     2
#define LFAST  2048
#define TSLOW  512
#define DFULL  1024
#define DHALF  512
#define NHEAD  8
#define HDIM   64
#define PHID   2048

typedef __attribute__((ext_vector_type(16))) __bf16 v16bf;
typedef __attribute__((ext_vector_type(8)))  __bf16 v8bf;
typedef __attribute__((ext_vector_type(8)))  float  v8f;
typedef __attribute__((ext_vector_type(4)))  int    v4i;

#define WMMA_BF16(a,b,c) \
  __builtin_amdgcn_wmma_f32_16x16x32_bf16(false,(a),false,(b),(short)0,(c),false,false)

// ---- CDNA5 async global->LDS path (guarded; sync ds-store fallback) ----
#if defined(__has_builtin)
#if __has_builtin(__builtin_amdgcn_global_load_async_to_lds_b128)
#define HAS_ASYNC_LDS 1
#endif
#endif

#ifdef HAS_ASYNC_LDS
typedef __attribute__((address_space(1))) v4i* as1_v4i;
typedef __attribute__((address_space(3))) v4i* as3_v4i;
#if __has_builtin(__builtin_amdgcn_s_wait_asynccnt)
#define WAIT_ASYNC(n) __builtin_amdgcn_s_wait_asynccnt(n)
#else
#define WAIT_ASYNC(n) asm volatile("s_wait_asynccnt %0" :: "i"(n) : "memory")
#endif
#else
#define WAIT_ASYNC(n)
#endif

__device__ __forceinline__ v8f zero8(){
  v8f z;
#pragma unroll
  for (int i=0;i<8;++i) z[i]=0.0f;
  return z;
}

// CDNA5 16-bit A/B fragment for one 16x32 (or 32x16) tile:
// lane holds row (A) / col (B) = lane&15; K-quadrant sel = (lane>>4)*8.
// elements [0..7]  = K = sel+0..7      -> p[0..7]
// elements [8..15] = K = 16+sel+0..7   -> p[16..23]
__device__ __forceinline__ v16bf load_frag16(const __bf16* p){
  v8bf lo = *(const v8bf*)(p);
  v8bf hi = *(const v8bf*)(p + 16);
  v16bf r;
#pragma unroll
  for (int i=0;i<8;++i){ r[i]=lo[i]; r[i+8]=hi[i]; }
  return r;
}

__device__ __forceinline__ float gelu_tanh(float x){
  float u = 0.7978845608028654f*(x + 0.044715f*x*x*x);
  return 0.5f*x*(1.0f + tanhf(u));
}

// ---------------- prep kernels ----------------
__global__ void k_f32_to_bf16(const float* __restrict__ in, __bf16* __restrict__ out, int n){
  int i = blockIdx.x*blockDim.x + threadIdx.x;
  if (i < n) out[i] = (__bf16)in[i];
}

// W: f32 [K][ldw], take Ncols starting at coff -> out bf16 [N][K] (row n, contiguous K)
__global__ void k_transpose_w(const float* __restrict__ w, int K, int ldw, int coff,
                              __bf16* __restrict__ out, int total){
  int i = blockIdx.x*blockDim.x + threadIdx.x;
  if (i >= total) return;
  int n = i / K, k = i - n*K;
  out[i] = (__bf16)w[(size_t)k*(size_t)ldw + coff + n];
}

// ybf[b, t, :] = ysmall[b, t/4, :]
__global__ void k_repeat4(const __bf16* __restrict__ in, __bf16* __restrict__ out, int n){
  int i = blockIdx.x*blockDim.x + threadIdx.x;
  if (i >= n) return;
  int row = i >> 9, col = i & 511;         // DHALF = 512
  int b = row >> 11, t = row & 2047;       // LFAST = 2048
  out[i] = in[((size_t)(b*TSLOW + (t>>2)))*DHALF + col];
}

// sshift[b,0,:]=0 ; sshift[b,t,:]=s[b,t-1,:]
__global__ void k_shift_rows(const __bf16* __restrict__ in, __bf16* __restrict__ out, int n){
  int i = blockIdx.x*blockDim.x + threadIdx.x;
  if (i >= n) return;
  int r = i & (TSLOW*DHALF - 1);
  int t = r >> 9;
  out[i] = (t == 0) ? (__bf16)0.0f : in[i - DHALF];
}

// ---------------- WMMA GEMM with async-LDS B staging ----------------
// C[M,N] = A[M,K](bf16, lda) * Bt[N,K](bf16, ldb)^T + bias
// grid = (N/64, M/128), block = (32,4). Each block: 128x64 tile.
// Wave w: rows [tm*128 + w*32, +32) x 64 cols -> 8 accumulators, 8 WMMA/k-step.
// B tile (64 rows x 32 k = 4KB) is shared by all 4 waves: double-buffered in LDS,
// streamed with GLOBAL_LOAD_ASYNC_TO_LDS_B128 (ASYNCcnt) when available.
#define GF_BIAS  1
#define GF_GELU  2
#define GF_BF16  4
#define GF_TRANS 8

__global__ __launch_bounds__(128) void k_gemm_bf16(
    const __bf16* __restrict__ A, int lda,
    const __bf16* __restrict__ Bt, int ldb,
    const float* __restrict__ bias,
    void* __restrict__ out, int ldo,
    int K, int flags)
{
  __shared__ __bf16 Bsh[2][64*32];   // 2 x 4KB

  const int lane = threadIdx.x;
  const int w    = threadIdx.y;
  const int tid  = w*32 + lane;
  const int m15  = lane & 15;
  const int sel  = (lane >> 4) * 8;
  const int tm   = blockIdx.y;
  const int tn   = blockIdx.x;

  // stage one 64x32 B tile (row-major, row stride 32 elems) into Bsh[buf]:
  // 256 16-byte segments; each of the 128 threads moves 2 segments.
  auto stage = [&](int buf, int kb){
#pragma unroll
    for (int p = 0; p < 2; ++p){
      int s = tid + p*128;
      const __bf16* src = Bt + (size_t)(tn*64 + (s>>2))*(size_t)ldb + kb + (s&3)*8;
      __bf16* dst = &Bsh[buf][s*8];
#ifdef HAS_ASYNC_LDS
      __builtin_amdgcn_global_load_async_to_lds_b128(
          (as1_v4i)(void*)(const_cast<__bf16*>(src)),
          (as3_v4i)(void*)(dst), 0, 0);
#else
      *(v8bf*)dst = *(const v8bf*)src;
#endif
    }
  };

  v8f acc[2][4];
#pragma unroll
  for (int i=0;i<2;++i)
#pragma unroll
    for (int j=0;j<4;++j) acc[i][j] = zero8();

  const __bf16* ap0 = A + (size_t)(tm*128 + w*32 + m15)*(size_t)lda + sel;
  const __bf16* ap1 = ap0 + (size_t)16*(size_t)lda;

  stage(0, 0);
  int ibuf = 0;
  for (int kb = 0; kb < K; kb += 32, ibuf ^= 1){
    const bool more = (kb + 32) < K;
    if (more) stage(ibuf ^ 1, kb + 32);     // prefetch next tile into other buffer
    if (more) { WAIT_ASYNC(2); } else { WAIT_ASYNC(0); }  // current tile's loads done
    __syncthreads();                         // all waves' stages visible

    v16bf a0 = load_frag16(ap0 + kb);        // global b128 loads
    v16bf a1 = load_frag16(ap1 + kb);
#pragma unroll
    for (int j=0;j<4;++j){
      v16bf bf = load_frag16(&Bsh[ibuf][(j*16 + m15)*32 + sel]);  // ds_load_b128
      acc[0][j] = WMMA_BF16(a0, bf, acc[0][j]);
      acc[1][j] = WMMA_BF16(a1, bf, acc[1][j]);
    }
    __syncthreads();                         // done reading before buffer reuse
  }

#pragma unroll
  for (int i=0;i<2;++i){
#pragma unroll
    for (int j=0;j<4;++j){
      int col = tn*64 + j*16 + m15;
      float bv = (flags & GF_BIAS) ? bias[col] : 0.0f;
#pragma unroll
      for (int r=0;r<8;++r){
        int row = tm*128 + w*32 + i*16 + sel + r;   // C layout: row = r + 8*(lane>>4)
        float v = acc[i][j][r] + bv;
        if (flags & GF_GELU) v = gelu_tanh(v);
        if (flags & GF_TRANS)      ((__bf16*)out)[(size_t)col*(size_t)ldo + row] = (__bf16)v;
        else if (flags & GF_BF16)  ((__bf16*)out)[(size_t)row*(size_t)ldo + col] = (__bf16)v;
        else                       ((float*) out)[(size_t)row*(size_t)ldo + col] = v;
      }
    }
  }
}

// ---------------- flash attention (wave32 WMMA, hd=64) ----------------
// Q,Kc: bf16 [B*T][512]; Vt: bf16 [512][B*Tk] (row d, ld=ldv); O: bf16 [B*Tq][512]
// grid = (Tq/64, NHEAD, B), block = (32,4): 1 wave = one 16-row q-tile of one (b,h).
__global__ __launch_bounds__(128) void k_flash_attn(
    const __bf16* __restrict__ Q,
    const __bf16* __restrict__ Kc,
    const __bf16* __restrict__ Vt, int ldv,
    __bf16* __restrict__ O,
    int Tq, int Tk)
{
  __shared__ float lsP[4][16*32];
  float* P = lsP[threadIdx.y];

  const int lane = threadIdx.x;
  const int m15  = lane & 15;
  const int sel  = (lane >> 4) * 8;
  const int h    = blockIdx.y;
  const int b    = blockIdx.z;
  const int qt   = blockIdx.x*4 + threadIdx.y;

  const __bf16* qp = Q + (size_t)(b*Tq + qt*16 + m15)*512 + h*HDIM + sel;
  v16bf qf0 = load_frag16(qp);        // d = 0..31
  v16bf qf1 = load_frag16(qp + 32);   // d = 32..63

  v8f o0 = zero8(), o1 = zero8(), o2 = zero8(), o3 = zero8();
  float mrow[8], lrow[8];
#pragma unroll
  for (int r=0;r<8;++r){ mrow[r] = -INFINITY; lrow[r] = 0.0f; }

  for (int t = 0; t < Tk; t += 32){
    // S(16x32) = Q(16x64) @ K_tile^T : B-operand rows are K[t..][d] directly.
    const __bf16* kp = Kc + (size_t)(b*Tk + t + m15)*512 + h*HDIM + sel;
    v8f s0 = zero8(), s1 = zero8();
    s0 = WMMA_BF16(qf0, load_frag16(kp),                       s0);
    s0 = WMMA_BF16(qf1, load_frag16(kp + 32),                  s0);
    s1 = WMMA_BF16(qf0, load_frag16(kp + (size_t)16*512),      s1);
    s1 = WMMA_BF16(qf1, load_frag16(kp + (size_t)16*512 + 32), s1);

    // online softmax across the 32 new columns; row r lives in VGPR r of each
    // accumulator across the 16 lanes of this half-wave.
    float alpha[8];
#pragma unroll
    for (int r=0;r<8;++r){
      float v0 = s0[r]*0.125f;   // 1/sqrt(64)
      float v1 = s1[r]*0.125f;
      float mx = fmaxf(v0, v1);
#pragma unroll
      for (int off=1; off<16; off<<=1)
        mx = fmaxf(mx, __shfl_xor(mx, off, 32));
      float mnew = fmaxf(mrow[r], mx);
      float p0 = __expf(v0 - mnew);
      float p1 = __expf(v1 - mnew);
      alpha[r] = __expf(mrow[r] - mnew);
      mrow[r]  = mnew;
      float rs = p0 + p1;
#pragma unroll
      for (int off=1; off<16; off<<=1)
        rs += __shfl_xor(rs, off, 32);
      lrow[r] = lrow[r]*alpha[r] + rs;
      s0[r] = p0; s1[r] = p1;
    }
#pragma unroll
    for (int r=0;r<8;++r){
      o0[r]*=alpha[r]; o1[r]*=alpha[r]; o2[r]*=alpha[r]; o3[r]*=alpha[r];
    }

    // C-layout -> A-fragment relayout of P through this wave's private LDS tile.
#pragma unroll
    for (int r=0;r<8;++r){
      P[(sel + r)*32 + m15]      = s0[r];
      P[(sel + r)*32 + 16 + m15] = s1[r];
    }
    __builtin_amdgcn_wave_barrier();
    v16bf pf;
#pragma unroll
    for (int i=0;i<8;++i){
      pf[i]   = (__bf16)P[m15*32 + sel + i];
      pf[i+8] = (__bf16)P[m15*32 + 16 + sel + i];
    }
    __builtin_amdgcn_wave_barrier();

    // O(16x64) += P(16x32) @ V_tile(32x64); B-operand rows come from Vt (contiguous in t).
    const __bf16* vp = Vt + (size_t)(h*HDIM + m15)*(size_t)ldv + b*Tk + t + sel;
    o0 = WMMA_BF16(pf, load_frag16(vp),                       o0);
    o1 = WMMA_BF16(pf, load_frag16(vp + (size_t)16*ldv),      o1);
    o2 = WMMA_BF16(pf, load_frag16(vp + (size_t)32*ldv),      o2);
    o3 = WMMA_BF16(pf, load_frag16(vp + (size_t)48*ldv),      o3);
  }

  __bf16* op = O + (size_t)(b*Tq + qt*16 + sel)*512 + h*HDIM + m15;
#pragma unroll
  for (int r=0;r<8;++r){
    float inv = 1.0f / lrow[r];
    op[(size_t)r*512 +  0] = (__bf16)(o0[r]*inv);
    op[(size_t)r*512 + 16] = (__bf16)(o1[r]*inv);
    op[(size_t)r*512 + 32] = (__bf16)(o2[r]*inv);
    op[(size_t)r*512 + 48] = (__bf16)(o3[r]*inv);
  }
}

// ---------------- host orchestration ----------------
extern "C" void kernel_launch(void* const* d_in, const int* in_sizes, int n_in,
                              void* d_out, int out_size, void* d_ws, size_t ws_size,
                              hipStream_t stream) {
  (void)in_sizes; (void)n_in; (void)out_size; (void)ws_size;

  const float* x_fast = (const float*)d_in[0];
  const float* x_slow = (const float*)d_in[1];
  const float* wqkv[4]; const float* bqkv[4]; const float* wo[4]; const float* bo[4];
  for (int i=0;i<4;++i){
    wqkv[i] = (const float*)d_in[2+4*i];
    bqkv[i] = (const float*)d_in[3+4*i];
    wo[i]   = (const float*)d_in[4+4*i];
    bo[i]   = (const float*)d_in[5+4*i];
  }
  const float* lift_w  = (const float*)d_in[18];
  const float* lift_b  = (const float*)d_in[19];
  const float* pool_w1 = (const float*)d_in[20];
  const float* pool_b1 = (const float*)d_in[21];
  const float* pool_w2 = (const float*)d_in[22];
  const float* pool_b2 = (const float*)d_in[23];

  // bump allocator over d_ws
  char* wsp = (char*)d_ws;
  auto alloc = [&](size_t nbytes)->void*{
    void* p = (void*)wsp;
    wsp += (nbytes + 255) & ~(size_t)255;
    return p;
  };

  const int MF = BB*LFAST;   // 4096 fast rows
  const int MS = BB*TSLOW;   // 1024 slow rows

  __bf16* xfast_bf = (__bf16*)alloc((size_t)MF*DFULL*2);
  __bf16* xslow_bf = (__bf16*)alloc((size_t)MS*DFULL*2);
  __bf16 *WqT[4], *WkT[4], *WvT[4], *WoT[4];
  for (int i=0;i<4;++i){
    WqT[i] = (__bf16*)alloc((size_t)DHALF*DHALF*2);
    WkT[i] = (__bf16*)alloc((size_t)DHALF*DHALF*2);
    WvT[i] = (__bf16*)alloc((size_t)DHALF*DHALF*2);
    WoT[i] = (__bf16*)alloc((size_t)DHALF*DHALF*2);
  }
  __bf16* liftT = (__bf16*)alloc((size_t)DHALF*DFULL*2);
  __bf16* pw1T  = (__bf16*)alloc((size_t)PHID*(4*DFULL)*2);
  __bf16* pw2T  = (__bf16*)alloc((size_t)DHALF*PHID*2);
  __bf16* ysm   = (__bf16*)alloc((size_t)MS*DHALF*2);
  __bf16* ybf   = (__bf16*)alloc((size_t)MF*DHALF*2);
  __bf16* hbf   = (__bf16*)alloc((size_t)MS*PHID*2);
  __bf16* sraw  = (__bf16*)alloc((size_t)MS*DHALF*2);
  __bf16* sshf  = (__bf16*)alloc((size_t)MS*DHALF*2);
  __bf16* Qbf   = (__bf16*)alloc((size_t)MF*DHALF*2);
  __bf16* Kbf   = (__bf16*)alloc((size_t)MF*DHALF*2);
  __bf16* Vt2   = (__bf16*)alloc((size_t)DHALF*MF*2);
  __bf16* Obf   = (__bf16*)alloc((size_t)MF*DHALF*2);

  auto gemm = [&](const __bf16* A, int lda, const __bf16* Bt, int ldb,
                  const float* bias, void* out, int ldo,
                  int M, int N, int K, int flags){
    dim3 g(N/64, M/128);
    k_gemm_bf16<<<g, dim3(32,4), 0, stream>>>(A, lda, Bt, ldb, bias, out, ldo, K, flags);
  };

  // ---- convert activations ----
  { int n = MF*DFULL; k_f32_to_bf16<<<n/256, 256, 0, stream>>>(x_fast, xfast_bf, n); }
  { int n = MS*DFULL; k_f32_to_bf16<<<n/256, 256, 0, stream>>>(x_slow, xslow_bf, n); }

  // ---- transpose + convert weights ----
  for (int i=0;i<4;++i){
    int tot = DHALF*DHALF;
    k_transpose_w<<<tot/256, 256, 0, stream>>>(wqkv[i], DHALF, 3*DHALF, 0,       WqT[i], tot);
    k_transpose_w<<<tot/256, 256, 0, stream>>>(wqkv[i], DHALF, 3*DHALF, DHALF,   WkT[i], tot);
    k_transpose_w<<<tot/256, 256, 0, stream>>>(wqkv[i], DHALF, 3*DHALF, 2*DHALF, WvT[i], tot);
    k_transpose_w<<<tot/256, 256, 0, stream>>>(wo[i],   DHALF, DHALF,   0,       WoT[i], tot);
  }
  { int tot = DHALF*DFULL;    k_transpose_w<<<tot/256,256,0,stream>>>(lift_w,  DFULL,   DHALF, 0, liftT, tot); }
  { int tot = PHID*(4*DFULL); k_transpose_w<<<tot/256,256,0,stream>>>(pool_w1, 4*DFULL, PHID,  0, pw1T,  tot); }
  { int tot = DHALF*PHID;     k_transpose_w<<<tot/256,256,0,stream>>>(pool_w2, PHID,    DHALF, 0, pw2T,  tot); }

  // ---- lift + repeat: y = repeat(x_slow @ lift_w + b, 4) ----
  gemm(xslow_bf, DFULL, liftT, DFULL, lift_b, ysm, DHALF, MS, DHALF, DFULL, GF_BIAS|GF_BF16);
  { int n = MF*DHALF; k_repeat4<<<n/256, 256, 0, stream>>>(ysm, ybf, n); }

  // ---- pool path: s = gelu(blocks@W1+b1)@W2+b2, then shift ----
  gemm(xfast_bf, 4*DFULL, pw1T, 4*DFULL, pool_b1, hbf, PHID, MS, PHID, 4*DFULL, GF_BIAS|GF_GELU|GF_BF16);
  gemm(hbf, PHID, pw2T, PHID, pool_b2, sraw, DHALF, MS, DHALF, PHID, GF_BIAS|GF_BF16);
  { int n = MS*DHALF; k_shift_rows<<<n/256, 256, 0, stream>>>(sraw, sshf, n); }

  float* z_fast = (float*)d_out;                         // [B*LFAST][1024]
  float* z_slow = (float*)d_out + (size_t)MF*DFULL;      // [B*TSLOW][1024]

  auto attn_block = [&](const __bf16* qA, int ldq, const __bf16* kvA, int ldkv,
                        int Tq, int Tk, int wi, float* zout){
    int Mq = BB*Tq, Mk = BB*Tk;
    gemm(qA,  ldq,  WqT[wi], DHALF, bqkv[wi],           Qbf, DHALF, Mq, DHALF, DHALF, GF_BIAS|GF_BF16);
    gemm(kvA, ldkv, WkT[wi], DHALF, bqkv[wi] + DHALF,   Kbf, DHALF, Mk, DHALF, DHALF, GF_BIAS|GF_BF16);
    gemm(kvA, ldkv, WvT[wi], DHALF, bqkv[wi] + 2*DHALF, Vt2, Mk,    Mk, DHALF, DHALF, GF_BIAS|GF_BF16|GF_TRANS);
    dim3 g(Tq/64, NHEAD, BB);
    k_flash_attn<<<g, dim3(32,4), 0, stream>>>(Qbf, Kbf, Vt2, Mk, Obf, Tq, Tk);
    gemm(Obf, DHALF, WoT[wi], DHALF, bo[wi], zout, DFULL, Mq, DHALF, DHALF, GF_BIAS);
  };

  // fast self:  x1 attends x1 -> z_fast[:, :512]
  attn_block(xfast_bf + 0,     DFULL, xfast_bf + 0, DFULL, LFAST, LFAST, 0, z_fast + 0);
  // fast cross: x2 attends y  -> z_fast[:, 512:]
  attn_block(xfast_bf + DHALF, DFULL, ybf,          DHALF, LFAST, LFAST, 1, z_fast + DHALF);
  // slow self:  s1 attends s1 -> z_slow[:, :512]
  attn_block(xslow_bf + 0,     DFULL, xslow_bf + 0, DFULL, TSLOW, TSLOW, 2, z_slow + 0);
  // slow cross: s2 attends shifted s -> z_slow[:, 512:]
  attn_block(xslow_bf + DHALF, DFULL, sshf,         DHALF, TSLOW, TSLOW, 3, z_slow + DHALF);
}
